// Decoder_46557445489302
// MI455X (gfx1250) — matrix operand order repaired and checked by reading, not compiled, not run on hardware
//
#include <hip/hip_runtime.h>
#include <math.h>

#define VOCAB 50257
#define HID   1024
#define EMB   1024
#define BOUND 25
#define G3    3072          // 3*HID
#define NTILES_OUT 3142     // ceil(VOCAB/16)
#define NBLK_OUT   393      // ceil(3142/8)
#define NPART      NBLK_OUT

typedef __bf16 bf16x4  __attribute__((ext_vector_type(4)));
typedef __bf16 bf16x8  __attribute__((ext_vector_type(8)));
typedef __bf16 bf16x16 __attribute__((ext_vector_type(16)));
typedef float  f32x8   __attribute__((ext_vector_type(8)));

// ---------------- workspace layout (bytes) ----------------
#define OFF_WOUT  ((size_t)0)                             // VOCAB*HID bf16 = 102,926,336
#define OFF_WIH   (OFF_WOUT + (size_t)VOCAB*HID*2)        // G3*HID bf16   = 6,291,456
#define OFF_WHH   (OFF_WIH  + (size_t)G3*HID*2)
#define OFF_H     (OFF_WHH  + (size_t)G3*HID*2)           // 1024 f32
#define OFF_HBF   (OFF_H    + 4096)                       // 1024 bf16
#define OFF_XBF   (OFF_HBF  + 2048)                       // 1024 bf16
#define OFF_GX    (OFF_XBF  + 2048)                       // 3072 f32
#define OFF_GH    (OFF_GX   + 12288)                      // 3072 f32
#define OFF_PVAL  (OFF_GH   + 12288)                      // 512 f32
#define OFF_PIDX  (OFF_PVAL + 2048)                       // 512 i32
#define OFF_FLAGS (OFF_PIDX + 2048)                       // [0]=word, [1]=done

// ---------------- f32 -> bf16 weight conversion (one-time per call) ----------------
__global__ void convert_weights(const float* __restrict__ Wout,
                                const float* __restrict__ Wih,
                                const float* __restrict__ Whh,
                                __bf16* __restrict__ WoutB,
                                __bf16* __restrict__ WihB,
                                __bf16* __restrict__ WhhB) {
  const size_t stride = (size_t)gridDim.x * blockDim.x;
  const size_t tid = (size_t)blockIdx.x * blockDim.x + threadIdx.x;
  const size_t N1 = (size_t)VOCAB * HID / 4;
  const size_t N2 = (size_t)G3 * HID / 4;
  for (size_t i = tid; i < N1; i += stride) {
    float4 v = ((const float4*)Wout)[i];
    bf16x4 o; o[0]=(__bf16)v.x; o[1]=(__bf16)v.y; o[2]=(__bf16)v.z; o[3]=(__bf16)v.w;
    ((bf16x4*)WoutB)[i] = o;
  }
  for (size_t i = tid; i < N2; i += stride) {
    float4 v = ((const float4*)Wih)[i];
    bf16x4 o; o[0]=(__bf16)v.x; o[1]=(__bf16)v.y; o[2]=(__bf16)v.z; o[3]=(__bf16)v.w;
    ((bf16x4*)WihB)[i] = o;
  }
  for (size_t i = tid; i < N2; i += stride) {
    float4 v = ((const float4*)Whh)[i];
    bf16x4 o; o[0]=(__bf16)v.x; o[1]=(__bf16)v.y; o[2]=(__bf16)v.z; o[3]=(__bf16)v.w;
    ((bf16x4*)WhhB)[i] = o;
  }
}

// ---------------- init: h, word, done, x=emb[sos] ----------------
__global__ void init_state(const float* __restrict__ hidden,
                           const float* __restrict__ embedding,
                           const int* __restrict__ sos_ptr,
                           float* __restrict__ h, __bf16* __restrict__ hbf,
                           __bf16* __restrict__ xbf, int* __restrict__ flags) {
  const int sos = *sos_ptr;
  if (threadIdx.x == 0) { flags[0] = sos; flags[1] = 0; }
  const float* erow = embedding + (size_t)sos * EMB;
  for (int j = threadIdx.x; j < HID; j += blockDim.x) {
    float hv = hidden[j];
    h[j] = hv;
    hbf[j] = (__bf16)hv;
    xbf[j] = (__bf16)erow[j];
  }
}

// ---------------- WMMA matvec tile: out[m0..m0+15] = W[tile] @ vec ----------------
// A: 16x32 bf16 tile of W (row = m0 + lane&15; lanes>=16 take the +8 K-chunks)
// B: vec broadcast to all 16 N columns (same K order as A; lanes>=16 take the +8 K-chunks)
// D: f32 16x16; lane 0 holds rows m0+v, lane 16 holds rows m0+8+v (all N equal)
__device__ __forceinline__ f32x8 gemv_tile_bf16(const __bf16* __restrict__ W,
                                                const __bf16* __restrict__ vec,
                                                int m0, int rows, int lane) {
  const int hi = (lane >> 4) & 1;
  int mrow = m0 + (lane & 15);
  if (mrow >= rows) mrow = rows - 1;               // clamp OOB loads (store is guarded)
  const __bf16* wrow = W + (size_t)mrow * HID;
  const int ka = hi ? 8 : 0;                       // A per-lane K offset
  const int kb = hi ? 8 : 0;                       // B per-lane K offset
  f32x8 c = {};
#pragma unroll 4
  for (int k0 = 0; k0 < HID; k0 += 32) {
    bf16x8 a_lo = *(const bf16x8*)(wrow + k0 + ka);
    bf16x8 a_hi = *(const bf16x8*)(wrow + k0 + 16 + ka);
    bf16x16 a = __builtin_shufflevector(a_lo, a_hi, 0,1,2,3,4,5,6,7,8,9,10,11,12,13,14,15);
    bf16x8 b_lo = *(const bf16x8*)(vec + k0 + kb);
    bf16x8 b_hi = *(const bf16x8*)(vec + k0 + 16 + kb);
    bf16x16 b = __builtin_shufflevector(b_lo, b_hi, 0,1,2,3,4,5,6,7,8,9,10,11,12,13,14,15);
    c = __builtin_amdgcn_wmma_f32_16x16x32_bf16(false, a, false, b, (short)0, c, false, false);
  }
  return c;
}

// ---------------- gates: gx = W_ih@x + b_ih ; gh = W_hh@h + b_hh ----------------
// 384 waves total (48 blocks x 8 waves): waves 0..191 -> gx, 192..383 -> gh
__global__ void gru_gates(const __bf16* __restrict__ WihB, const __bf16* __restrict__ WhhB,
                          const float* __restrict__ b_ih, const float* __restrict__ b_hh,
                          const __bf16* __restrict__ xbf, const __bf16* __restrict__ hbf,
                          float* __restrict__ gx, float* __restrict__ gh) {
  const int wave = (int)((blockIdx.x * blockDim.x + threadIdx.x) >> 5);
  const int lane = threadIdx.x & 31;
  const __bf16* W;  const __bf16* vec;  const float* bias;  float* out;  int m0;
  if (wave < 192) { W = WihB; vec = xbf; bias = b_ih; out = gx; m0 = wave * 16; }
  else            { W = WhhB; vec = hbf; bias = b_hh; out = gh; m0 = (wave - 192) * 16; }
  f32x8 c = gemv_tile_bf16(W, vec, m0, G3, lane);
  if ((lane & 15) == 0) {
    const int base = m0 + ((lane >> 4) ? 8 : 0);
#pragma unroll
    for (int v = 0; v < 8; ++v) out[base + v] = c[v] + bias[base + v];
  }
}

// ---------------- GRU pointwise update (PyTorch gate order r,z,n) ----------------
__global__ void gru_update(const float* __restrict__ gx, const float* __restrict__ gh,
                           float* __restrict__ h, __bf16* __restrict__ hbf,
                           const int* __restrict__ flags) {
  const int done = flags[1];
  for (int j = threadIdx.x; j < HID; j += blockDim.x) {
    float r = 1.f / (1.f + __expf(-(gx[j] + gh[j])));
    float z = 1.f / (1.f + __expf(-(gx[HID + j] + gh[HID + j])));
    float n = tanhf(gx[2 * HID + j] + r * gh[2 * HID + j]);
    float hn = (1.f - z) * n + z * h[j];
    float hv = done ? h[j] : hn;        // freeze h after EOS
    h[j] = hv;
    hbf[j] = (__bf16)hv;
  }
}

// ---------------- logits = W_out@h + b_out ; per-block argmax partial ----------------
__global__ void logits_argmax(const __bf16* __restrict__ WoutB, const float* __restrict__ b_out,
                              const __bf16* __restrict__ hbf,
                              float* __restrict__ step_logits, const int* __restrict__ flags,
                              float* __restrict__ pval, int* __restrict__ pidx) {
  __shared__ float sval[16];
  __shared__ int   sidx[16];
  const int waveInBlk = threadIdx.x >> 5;
  const int lane = threadIdx.x & 31;
  const int tile = (int)blockIdx.x * 8 + waveInBlk;
  float bestV = -INFINITY;
  int   bestI = 0x7fffffff;
  if (tile < NTILES_OUT) {
    const int m0 = tile * 16;
    f32x8 c = gemv_tile_bf16(WoutB, hbf, m0, VOCAB, lane);
    const int done = flags[1];
    if ((lane & 15) == 0) {
      const int base = m0 + ((lane >> 4) ? 8 : 0);
#pragma unroll
      for (int v = 0; v < 8; ++v) {
        const int m = base + v;
        if (m < VOCAB) {
          float val = c[v] + b_out[m];
          // streaming store: keep bf16 W_out resident in L2
          __builtin_nontemporal_store(done ? 0.f : val, &step_logits[m]);
          if (val > bestV || (val == bestV && m < bestI)) { bestV = val; bestI = m; }
        }
      }
    }
  }
  if ((lane & 15) == 0) {
    sval[waveInBlk * 2 + (lane >> 4)] = bestV;
    sidx[waveInBlk * 2 + (lane >> 4)] = bestI;
  }
  __syncthreads();
  if (threadIdx.x == 0) {
    float bv = -INFINITY; int bi = 0x7fffffff;
#pragma unroll
    for (int i = 0; i < 16; ++i) {
      float v = sval[i]; int ix = sidx[i];
      if (v > bv || (v == bv && ix < bi)) { bv = v; bi = ix; }
    }
    pval[blockIdx.x] = bv;
    pidx[blockIdx.x] = bi;
  }
}

// ---------------- final argmax + word/done update + next-step embedding to bf16 ----------------
__global__ void finalize_step(const float* __restrict__ pval, const int* __restrict__ pidx,
                              int* __restrict__ flags, int* __restrict__ words_out,
                              const int* __restrict__ eos_ptr,
                              const float* __restrict__ embedding, __bf16* __restrict__ xbf,
                              int t) {
  __shared__ float sv[512];
  __shared__ int   si[512];
  __shared__ int   sWord;
  const int tid = threadIdx.x;
  float bv = -INFINITY; int bi = 0x7fffffff;
  for (int i = tid; i < NPART; i += 512) {
    float v = pval[i]; int ix = pidx[i];
    if (v > bv || (v == bv && ix < bi)) { bv = v; bi = ix; }
  }
  sv[tid] = bv; si[tid] = bi;
  __syncthreads();
  for (int s = 256; s > 0; s >>= 1) {
    if (tid < s) {
      float v = sv[tid + s]; int ix = si[tid + s];
      if (v > sv[tid] || (v == sv[tid] && ix < si[tid])) { sv[tid] = v; si[tid] = ix; }
    }
    __syncthreads();
  }
  if (tid == 0) {
    const int idx = si[0];
    const int done = flags[1];
    const int eos = *eos_ptr;
    const int new_done = done | (idx == eos);
    words_out[t] = new_done ? -1 : idx;                 // emit_word
    const int wnext = new_done ? flags[0] : idx;        // word_next
    flags[0] = wnext;
    flags[1] = new_done;
    sWord = wnext;
  }
  __syncthreads();
  const float* erow = embedding + (size_t)sWord * EMB;  // prefetch next x in bf16
  for (int j = tid; j < EMB; j += 512) xbf[j] = (__bf16)erow[j];
}

// ---------------- host-side launch ----------------
extern "C" void kernel_launch(void* const* d_in, const int* in_sizes, int n_in,
                              void* d_out, int out_size, void* d_ws, size_t ws_size,
                              hipStream_t stream) {
  const float* hidden    = (const float*)d_in[0];
  const float* embedding = (const float*)d_in[1];
  const float* W_ih      = (const float*)d_in[2];
  const float* W_hh      = (const float*)d_in[3];
  const float* b_ih      = (const float*)d_in[4];
  const float* b_hh      = (const float*)d_in[5];
  const float* W_out     = (const float*)d_in[6];
  const float* b_out     = (const float*)d_in[7];
  const int*   sos_ptr   = (const int*)d_in[8];
  const int*   eos_ptr   = (const int*)d_in[9];

  char* ws = (char*)d_ws;
  __bf16* WoutB = (__bf16*)(ws + OFF_WOUT);
  __bf16* WihB  = (__bf16*)(ws + OFF_WIH);
  __bf16* WhhB  = (__bf16*)(ws + OFF_WHH);
  float*  h     = (float*) (ws + OFF_H);
  __bf16* hbf   = (__bf16*)(ws + OFF_HBF);
  __bf16* xbf   = (__bf16*)(ws + OFF_XBF);
  float*  gx    = (float*) (ws + OFF_GX);
  float*  gh    = (float*) (ws + OFF_GH);
  float*  pval  = (float*) (ws + OFF_PVAL);
  int*    pidx  = (int*)   (ws + OFF_PIDX);
  int*    flags = (int*)   (ws + OFF_FLAGS);

  // d_out: words[BOUND] as int32 bit patterns, then logits[BOUND][VOCAB] f32
  int*   words_out  = (int*)d_out;
  float* logits_out = (float*)d_out + BOUND;

  // one-time per call: bf16 weight images (W_out fits in the 192MB L2 as bf16)
  convert_weights<<<2048, 256, 0, stream>>>(W_out, W_ih, W_hh, WoutB, WihB, WhhB);
  init_state<<<1, 256, 0, stream>>>(hidden, embedding, sos_ptr, h, hbf, xbf, flags);

  for (int t = 0; t < BOUND; ++t) {
    gru_gates<<<48, 256, 0, stream>>>(WihB, WhhB, b_ih, b_hh, xbf, hbf, gx, gh);
    gru_update<<<1, 256, 0, stream>>>(gx, gh, h, hbf, flags);
    logits_argmax<<<NBLK_OUT, 256, 0, stream>>>(WoutB, b_out, hbf,
                                                logits_out + (size_t)t * VOCAB,
                                                flags, pval, pidx);
    finalize_step<<<1, 512, 0, stream>>>(pval, pidx, flags, words_out, eos_ptr,
                                         embedding, xbf, t);
  }
}
